// MoE_47845935677787
// MI455X (gfx1250) — compile-verified, matrix operand order checked by reading, not run on hardware
//
#include <hip/hip_runtime.h>

// ---------------- problem constants (match reference) ----------------
#define BB   16
#define LL   2048
#define DD   128
#define EE   32
#define KK   8
#define DFF  512
#define TT   (BB * LL)      // 32768 tokens
#define TILE_M 256
#define NTHREADS 512        // 16 waves

typedef __attribute__((ext_vector_type(16))) __bf16 v16bf;
typedef __attribute__((ext_vector_type(8)))  __bf16 v8bf;
typedef __attribute__((ext_vector_type(8)))  float  v8f;
typedef __attribute__((ext_vector_type(4)))  int    v4i;

// ---------------- async global->LDS copy (gfx1250 ASYNCcnt path) --------
#if defined(__has_builtin)
#if __has_builtin(__builtin_amdgcn_global_load_async_to_lds_b128) && \
    __has_builtin(__builtin_amdgcn_s_wait_asynccnt)
#define HAVE_ASYNC 1
#endif
#endif

typedef __attribute__((address_space(1))) v4i gv4i;   // global (AS1) int4
typedef __attribute__((address_space(3))) v4i lv4i;   // LDS    (AS3) int4

__device__ __forceinline__ void async_copy16(__bf16* dst_lds, const __bf16* src_g) {
#if HAVE_ASYNC
  __builtin_amdgcn_global_load_async_to_lds_b128(
      (gv4i*)(void*)src_g, (lv4i*)(void*)dst_lds, 0, 0);
#else
  *(v8bf*)dst_lds = *(const v8bf*)src_g;
#endif
}

__device__ __forceinline__ void wait_lds_copies() {
#if HAVE_ASYNC
  __builtin_amdgcn_s_wait_asynccnt(0);
#endif
}

// Load a 16x32 bf16 WMMA operand tile (A layout; B uses the same layout on
// the transposed matrix) from a row-major LDS array.
// Per ISA 7.12.2: lane m (0..15) holds K = {0..7, 16..23}, lane m+16 holds
// K = {8..15, 24..31} of row m -> two 16-byte LDS loads per lane.
__device__ __forceinline__ v16bf load_tile(const __bf16* base, int stride,
                                           int rbase, int kbase, int lane) {
  int r    = rbase + (lane & 15);
  int koff = (lane >> 4) << 3;          // 0 or 8
  const __bf16* p = base + r * stride + kbase + koff;
  union { v16bf v; v8bf h[2]; } u;
  u.h[0] = *(const v8bf*)(p);           // K = koff .. koff+7
  u.h[1] = *(const v8bf*)(p + 16);      // K = 16+koff .. 16+koff+7
  return u.v;
}

__device__ __forceinline__ v8f vzero8() {
  v8f z = {0.f, 0.f, 0.f, 0.f, 0.f, 0.f, 0.f, 0.f};
  return z;
}

// ---------------- kernel 0: zero output + expert counts ----------------
__global__ void moe_init_kernel(float* __restrict__ out, int n,
                                int* __restrict__ counts) {
  int i = blockIdx.x * blockDim.x + threadIdx.x;
  if (i < n) out[i] = 0.f;
  if (i < EE) counts[i] = 0;
}

// ---------------- prep kernels: fp32 -> bf16 (and W transposes) ----------
__global__ void prep_x_kernel(const float* __restrict__ x,
                              __bf16* __restrict__ xbf, int n) {
  int i = blockIdx.x * blockDim.x + threadIdx.x;
  if (i < n) xbf[i] = (__bf16)x[i];
}

// W1T layout: [e][f(DFF)][d(D)]  (so a 32-wide DFF chunk is 8KB contiguous)
__global__ void prep_w1_kernel(const float* __restrict__ W1,
                               __bf16* __restrict__ W1T) {
  int idx = blockIdx.x * blockDim.x + threadIdx.x;   // over E*DFF*DD
  int d = idx & (DD - 1);
  int f = (idx >> 7) & (DFF - 1);
  int e = idx >> 16;
  W1T[idx] = (__bf16)W1[((size_t)e * DD + d) * DFF + f];
}

// W2T layout: [e][d(D)][f(DFF)]  (B-operand rows along DFF)
__global__ void prep_w2_kernel(const float* __restrict__ W2,
                               __bf16* __restrict__ W2T) {
  int idx = blockIdx.x * blockDim.x + threadIdx.x;   // over E*DD*DFF
  int f = idx & (DFF - 1);
  int d = (idx >> 9) & (DD - 1);
  int e = idx >> 16;
  W2T[idx] = (__bf16)W2[((size_t)e * DFF + f) * DD + d];
}

// ---------------- kernel 1: router (one wave per token, lane = expert) ----
__global__ void moe_router_kernel(const float* __restrict__ x,
                                  const float* __restrict__ Wr,
                                  const float* __restrict__ br,
                                  int* __restrict__ k_ids,
                                  float* __restrict__ k_w,
                                  int* __restrict__ counts) {
  __shared__ float WrS[EE * DD];                       // 16 KB
  int tid = threadIdx.x;
  for (int i = tid; i < EE * DD; i += blockDim.x) WrS[i] = Wr[i];
  __syncthreads();

  int wave = tid >> 5, lane = tid & 31;
  int t = blockIdx.x * 8 + wave;                       // 8 tokens / block
  const float* xr = x + (size_t)t * DD;

  float acc = br[lane];                                // lane == expert
  #pragma unroll 4
  for (int d = 0; d < DD; ++d) acc += xr[d] * WrS[lane * DD + d];

  // rank of this expert's logit among all 32 (descending, index tiebreak)
  float maxv = -3.4e38f;
  int rank = 0;
  for (int j = 0; j < 32; ++j) {
    float vj = __shfl(acc, j, 32);
    maxv = vj > maxv ? vj : maxv;
    if (vj > acc || (vj == acc && j < lane)) ++rank;
  }
  bool sel = rank < KK;
  float ev = sel ? __expf(acc - maxv) : 0.f;
  float s = ev;
  #pragma unroll
  for (int m = 16; m >= 1; m >>= 1) s += __shfl_xor(s, m, 32);

  if (sel) {
    k_ids[t * KK + rank] = lane;
    k_w[t * KK + rank]   = ev / s;                     // softmax over top-K
    atomicAdd(&counts[lane], 1);
  }
}

// ---------------- kernel 2: exclusive scan over 32 expert counts ----------
__global__ void moe_scan_kernel(const int* __restrict__ counts,
                                int* __restrict__ offsets,
                                int* __restrict__ cursors) {
  if (threadIdx.x == 0) {
    int acc = 0;
    for (int e = 0; e < EE; ++e) {
      offsets[e] = acc;
      cursors[e] = acc;
      acc += counts[e];
    }
  }
}

// ---------------- kernel 3: build per-expert gathered row lists -----------
__global__ void moe_build_kernel(const int* __restrict__ k_ids,
                                 const float* __restrict__ k_w,
                                 int* __restrict__ cursors,
                                 int* __restrict__ rows,
                                 float* __restrict__ rw) {
  int t = blockIdx.x * blockDim.x + threadIdx.x;
  if (t >= TT) return;
  #pragma unroll
  for (int k = 0; k < KK; ++k) {
    int e = k_ids[t * KK + k];
    int pos = atomicAdd(&cursors[e], 1);
    rows[pos] = t;
    rw[pos]   = k_w[t * KK + k];
  }
}

// ---------------- kernel 4: fused per-expert FFN (bf16 WMMA) --------------
// Block: 512 threads (16 waves), tile = 256 gathered tokens x full D=128 out.
// Streams DFF in chunks of 32 (one WMMA K-step):
//   H_c = relu(Xg @ W1[:, c]) ;  O += H_c @ W2[c, :]
// Weight chunks are double-buffered and staged with async global->LDS copies
// issued one chunk ahead, so staging overlaps WMMA compute.
__global__ __launch_bounds__(NTHREADS)
void moe_expert_kernel(const __bf16* __restrict__ xbf,
                       const __bf16* __restrict__ W1T,
                       const __bf16* __restrict__ W2T,
                       const float* __restrict__ b1,
                       const float* __restrict__ b2,
                       const int* __restrict__ counts,
                       const int* __restrict__ offsets,
                       const int* __restrict__ rows,
                       const float* __restrict__ rw,
                       float* __restrict__ out) {
  int e = blockIdx.y;
  int cnt = counts[e];
  int tile0 = blockIdx.x * TILE_M;
  if (tile0 >= cnt) return;
  int base = offsets[e];

  __shared__ __bf16 Xs[TILE_M * DD];     // gathered tokens, row-major   64 KB
  __shared__ __bf16 W1s[2][32 * DD];     // W1 chunk^T [n(dff)][k(D)] 2x 8 KB
  __shared__ __bf16 W2s[2][DD * 32];     // W2 chunk^T [n(D)][kc]     2x 8 KB
  __shared__ __bf16 Hs[TILE_M * 32];     // relu hidden chunk [m][kc]    16 KB
  __shared__ int   toks[TILE_M];
  __shared__ float tws[TILE_M];

  int tid = threadIdx.x, lane = tid & 31, wave = tid >> 5;

  if (tid < TILE_M) {
    int r = tile0 + tid;
    if (r < cnt) { toks[tid] = rows[base + r]; tws[tid] = rw[base + r]; }
    else         { toks[tid] = 0;              tws[tid] = 0.f; }
  }
  __syncthreads();

  const __bf16* W1e = W1T + (size_t)e * DFF * DD;   // [f][d]
  const __bf16* W2e = W2T + (size_t)e * DD * DFF;   // [d][f]
  const float*  b1e = b1 + (size_t)e * DFF;
  const float*  b2e = b2 + (size_t)e * DD;

  // async gather of the 256x128 bf16 activation tile (8x 16B per thread)
  for (int i = tid; i < TILE_M * 16; i += NTHREADS) {
    int m = i >> 4, j = (i & 15) << 3;
    async_copy16(&Xs[m * DD + j], xbf + (size_t)toks[m] * DD + j);
  }
  // stage chunk 0 weights (1x W1 + 1x W2 16B copy per thread)
  {
    async_copy16(&W1s[0][tid * 8], W1e + (size_t)0 * 32 * DD + tid * 8);
    int n = tid >> 2, q = (tid & 3) << 3;
    async_copy16(&W2s[0][n * 32 + q], W2e + (size_t)n * DFF + 0 + q);
  }

  v8f o[8];
  #pragma unroll
  for (int i = 0; i < 8; ++i) o[i] = vzero8();

  for (int c = 0; c < DFF / 32; ++c) {
    wait_lds_copies();      // my wave's chunk-c (and Xs) copies landed
    __syncthreads();        // everyone's landed; prev chunk's compute done
    int b = c & 1;

    if (c + 1 < DFF / 32) {   // prefetch next chunk into the other buffer
      int nb = (c + 1) & 1;
      async_copy16(&W1s[nb][tid * 8],
                   W1e + (size_t)(c + 1) * 32 * DD + tid * 8);
      int n = tid >> 2, q = (tid & 3) << 3;
      async_copy16(&W2s[nb][n * 32 + q],
                   W2e + (size_t)n * DFF + (c + 1) * 32 + q);
    }

    // ---- GEMM1: H_c[256x32] = Xs[256x128] @ W1chunk.  wave -> M-tile ----
    v8f h0 = vzero8(), h1 = vzero8();
    #pragma unroll
    for (int kk = 0; kk < DD; kk += 32) {
      v16bf a  = load_tile(Xs,     DD, wave * 16, kk, lane);
      v16bf bA = load_tile(W1s[b], DD, 0,         kk, lane);
      v16bf bB = load_tile(W1s[b], DD, 16,        kk, lane);
      h0 = __builtin_amdgcn_wmma_f32_16x16x32_bf16(false, a, false, bA,
                                                   (short)0, h0, false, false);
      h1 = __builtin_amdgcn_wmma_f32_16x16x32_bf16(false, a, false, bB,
                                                   (short)0, h1, false, false);
    }
    // bias + ReLU + bf16 -> Hs. Each wave writes (and later reads) only its
    // own 16 rows: per-wave LDS ordering makes this safe without a barrier.
    {
      int ncol = lane & 15;
      int moff = wave * 16 + ((lane >> 4) << 3);
      float bA = b1e[c * 32 + ncol];
      float bB = b1e[c * 32 + 16 + ncol];
      #pragma unroll
      for (int v = 0; v < 8; ++v) {
        float f0 = h0[v] + bA; f0 = f0 > 0.f ? f0 : 0.f;
        float f1 = h1[v] + bB; f1 = f1 > 0.f ? f1 : 0.f;
        int m = moff + v;
        Hs[m * 32 + ncol]      = (__bf16)f0;
        Hs[m * 32 + 16 + ncol] = (__bf16)f1;
      }
    }

    // ---- GEMM2: O[256x128] += Hs[256x32] @ W2chunk. wave -> M-tile, 8 N ---
    v16bf a2 = load_tile(Hs, 32, wave * 16, 0, lane);
    #pragma unroll
    for (int nt = 0; nt < 8; ++nt) {
      v16bf bT = load_tile(W2s[b], 32, nt * 16, 0, lane);
      o[nt] = __builtin_amdgcn_wmma_f32_16x16x32_bf16(false, a2, false, bT,
                                                      (short)0, o[nt], false, false);
    }
  }

  // ---- epilogue: gate-weighted scatter-add (includes b2) ----
  {
    int ncl  = lane & 15;
    int moff = wave * 16 + ((lane >> 4) << 3);
    #pragma unroll
    for (int nt = 0; nt < 8; ++nt) {
      float bb = b2e[nt * 16 + ncl];
      #pragma unroll
      for (int v = 0; v < 8; ++v) {
        int m = moff + v;
        if (tile0 + m < cnt) {
          float val = o[nt][v] + bb;
          atomicAdd(&out[(size_t)toks[m] * DD + nt * 16 + ncl], tws[m] * val);
        }
      }
    }
  }
}

// ---------------- host-side launch ----------------
extern "C" void kernel_launch(void* const* d_in, const int* in_sizes, int n_in,
                              void* d_out, int out_size, void* d_ws, size_t ws_size,
                              hipStream_t stream) {
  const float* x  = (const float*)d_in[0];
  const float* Wr = (const float*)d_in[1];
  const float* br = (const float*)d_in[2];
  const float* W1 = (const float*)d_in[3];
  const float* b1 = (const float*)d_in[4];
  const float* W2 = (const float*)d_in[5];
  const float* b2 = (const float*)d_in[6];
  float* out = (float*)d_out;

  // workspace carve (~21 MB), every section 16B aligned
  int*    counts  = (int*)d_ws;               // E
  int*    offsets = counts + EE;              // E
  int*    cursors = offsets + EE;             // E + 32 pad
  int*    k_ids   = cursors + EE + 32;        // T*K
  int*    rows    = k_ids + (size_t)TT * KK;  // T*K
  float*  k_w     = (float*)(rows + (size_t)TT * KK);   // T*K
  float*  rw      = k_w + (size_t)TT * KK;              // T*K
  __bf16* xbf     = (__bf16*)(rw + (size_t)TT * KK);    // T*D
  __bf16* W1T     = xbf + (size_t)TT * DD;              // E*DFF*D
  __bf16* W2T     = W1T + (size_t)EE * DFF * DD;        // E*D*DFF

  // 0) zero out + counts
  moe_init_kernel<<<(out_size + 255) / 256, 256, 0, stream>>>(out, out_size, counts);
  // prep: bf16 conversions / transposes (one-time ~50MB of HBM traffic)
  prep_x_kernel<<<(TT * DD + 255) / 256, 256, 0, stream>>>(x, xbf, TT * DD);
  prep_w1_kernel<<<(EE * DFF * DD) / 256, 256, 0, stream>>>(W1, W1T);
  prep_w2_kernel<<<(EE * DD * DFF) / 256, 256, 0, stream>>>(W2, W2T);
  // 1) router: 8 tokens per 256-thread block
  moe_router_kernel<<<TT / 8, 256, 0, stream>>>(x, Wr, br, k_ids, k_w, counts);
  // 2) scan
  moe_scan_kernel<<<1, 32, 0, stream>>>(counts, offsets, cursors);
  // 3) build gathered lists
  moe_build_kernel<<<TT / 256, 256, 0, stream>>>(k_ids, k_w, cursors, rows, rw);
  // 4) per-expert fused FFN; worst case 128 tiles of 256 rows per expert
  dim3 grid(TT / TILE_M, EE);
  moe_expert_kernel<<<grid, NTHREADS, 0, stream>>>(xbf, W1T, W2T, b1, b2,
                                                   counts, offsets, rows, rw, out);
}
